// attentive_pooling_52518860095628
// MI455X (gfx1250) — compile-verified, matrix operand order checked by reading, not run on hardware
//
#include <hip/hip_runtime.h>
#include <hip/hip_bf16.h>
#include <math.h>

// ---------------------------------------------------------------------------
// Attention with materialized probabilities, MI455X (gfx1250) wave32 + WMMA.
//   context  = softmax(Q K^T * scale) V        [B, Sq, D]
//   attention= softmax(Q K^T * scale)          [B, Sq, Sk]
// B=32, Sq=Sk=2048, D=128, fp32 in/out. Matmuls in bf16 WMMA w/ f32 accum.
// One workgroup = 16 query rows of one batch; full S row-block (16x2048 f32 =
// 128 KB) lives in LDS so the 512MB attention matrix is written exactly once.
// ---------------------------------------------------------------------------

typedef __attribute__((ext_vector_type(16))) __bf16 v16bf;
typedef __attribute__((ext_vector_type(8)))  float  v8f;

#define B_   32
#define SQ_  2048
#define SK_  2048
#define D_   128
#define QTILE 16                 // query rows per workgroup
#define NWAVES 8                 // 256 threads / wave32
#define SCALE_F 0.08838834764831845f   // 1/sqrt(128)

// 16-bit A-matrix 16x32 fragment coords (CDNA5 ISA 7.12.2):
//   lane 0-15 -> M = lane, lanes 16-31 -> same M, K offset +8
//   vgpr r (=i/2), half h (=i%2): K = (r>=4?16:0) + (r&3)*2 + h (+8 if lane>=16)
// B-matrix 32x16 uses the same pattern with N in place of M.
__device__ __forceinline__ void frag_coords(int lane, int i, int& mn, int& kk) {
    int r = i >> 1, h = i & 1;
    mn = lane & 15;
    kk = ((r & 4) << 2) + ((r & 3) << 1) + h + ((lane & 16) >> 1);
}

__global__ void
__launch_bounds__(256, 1)
attn_fwd_kernel(const float* __restrict__ qg,
                const float* __restrict__ kg,
                const float* __restrict__ vg,
                float* __restrict__ ctx_out,     // [B, Sq, D]
                float* __restrict__ attn_out)    // [B, Sq, Sk]
{
    extern __shared__ float smem[];
    float* Slds  = smem;                          // [QTILE][SK_]  128 KB
    float* red   = smem + QTILE * SK_;            // [16][16] partials
    float* rowm  = red + 256;                     // [16] row max (raw scores)
    float* rowis = rowm + 16;                     // [16] 1/rowsum

    const int tid   = threadIdx.x;
    const int lane  = tid & 31;
    const int wid   = tid >> 5;
    const int qb    = blockIdx.x;                 // query tile index
    const int b     = blockIdx.y;                 // batch
    const int qbase = qb * QTILE;

    // ---------------- Phase 1: Q tile -> bf16 A fragments (held in VGPRs) ---
    // A is 16x32 per WMMA; D=128 -> 4 K-chunks.
    v16bf aq[4];
    {
        const size_t qrow0 = ((size_t)b * SQ_ + qbase) * D_;
#pragma unroll
        for (int kc = 0; kc < 4; ++kc) {
#pragma unroll
            for (int i = 0; i < 16; ++i) {
                int m, kk; frag_coords(lane, i, m, kk);
                aq[kc][i] = (__bf16)qg[qrow0 + (size_t)m * D_ + kc * 32 + kk];
            }
        }
    }

    // ---------------- Phase 2: S = Q K^T into LDS (bf16 WMMA, f32 acc) ------
    // S tile is 16 x 2048 = 1 Mblock x 128 Nblocks of 16; wave w owns
    // Nblocks {w, w+8, ..., w+120}.
    {
        const size_t krow0 = (size_t)b * SK_ * D_;
#pragma unroll 1
        for (int t = 0; t < 16; ++t) {
            const int nb = wid + (t << 3);        // key block: rows nb*16..+15
            // prefetch next key tile for this wave (speculative, NT)
            if (t < 15) {
                const float* pf = kg + krow0 + (size_t)((nb + 8) * 16 + (lane & 15)) * D_;
                __builtin_prefetch(pf, 0, 1);
            }
            v8f acc = {};
#pragma unroll
            for (int kc = 0; kc < 4; ++kc) {
                v16bf bk;
#pragma unroll
                for (int i = 0; i < 16; ++i) {
                    int n, kk; frag_coords(lane, i, n, kk);
                    // B[k][n] = K[nb*16+n][kc*32+kk]   (K^T)
                    bk[i] = (__bf16)kg[krow0 + (size_t)(nb * 16 + n) * D_ + kc * 32 + kk];
                }
                acc = __builtin_amdgcn_wmma_f32_16x16x32_bf16(
                        false, aq[kc], false, bk, (short)0, acc, false, false);
            }
            // scatter C fragment (16x16 f32) into Slds
#pragma unroll
            for (int r = 0; r < 8; ++r) {
                const int m = (lane < 16) ? r : (r + 8);
                const int n = lane & 15;
                Slds[m * SK_ + nb * 16 + n] = acc[r];
            }
        }
    }
    __syncthreads();

    // ---------------- Phase 3: row softmax statistics in LDS ----------------
    // 256 threads = 16 rows x 16 segments of 128 columns.
    {
        const int row = tid >> 4;
        const int seg = tid & 15;
        float* srow = Slds + row * SK_ + seg * 128;
        float mx = -INFINITY;
#pragma unroll 4
        for (int c = 0; c < 128; ++c) mx = fmaxf(mx, srow[c]);
        red[tid] = mx;
        __syncthreads();
        if (seg == 0) {
            float m2 = red[row * 16];
#pragma unroll
            for (int i = 1; i < 16; ++i) m2 = fmaxf(m2, red[row * 16 + i]);
            rowm[row] = m2;
        }
        __syncthreads();

        const float rm = rowm[row];
        float sm = 0.f;
#pragma unroll 4
        for (int c = 0; c < 128; ++c) {
            float e = __expf((srow[c] - rm) * SCALE_F);
            srow[c] = e;                     // unnormalized p
            sm += e;
        }
        red[tid] = sm;
        __syncthreads();
        if (seg == 0) {
            float s2 = 0.f;
#pragma unroll
            for (int i = 0; i < 16; ++i) s2 += red[row * 16 + i];
            rowis[row] = 1.0f / s2;
        }
        __syncthreads();
    }

    // ---------------- Phase 4: normalize + stream P to HBM (coalesced) ------
    {
        const size_t abase = ((size_t)b * SQ_ + qbase) * (size_t)SK_;
#pragma unroll 1
        for (int it = 0; it < (QTILE * SK_) / 256; ++it) {   // 128 iterations
            const int idx = it * 256 + tid;                   // row*2048 + col
            const int r   = idx >> 11;
            float p = Slds[idx] * rowis[r];
            Slds[idx] = p;
            attn_out[abase + idx] = p;                        // contiguous
        }
    }
    __syncthreads();

    // ---------------- Phase 5: O = P V (bf16 WMMA over 2048 keys) -----------
    // O tile 16x128 = 8 Nblocks of 16; wave w owns Nblock w.
    {
        const size_t vrow0 = (size_t)b * SK_ * D_;
        const int nb = wid;                        // output cols nb*16..+15
        v8f oacc = {};
#pragma unroll 1
        for (int kc2 = 0; kc2 < SK_ / 32; ++kc2) { // 64 chunks of 32 keys
            v16bf ap, bv;
#pragma unroll
            for (int i = 0; i < 16; ++i) {
                int mn, kk; frag_coords(lane, i, mn, kk);
                // A = P[m][k]  from LDS
                ap[i] = (__bf16)Slds[mn * SK_ + kc2 * 32 + kk];
                // B = V[k][n]  from global
                bv[i] = (__bf16)vg[vrow0 + (size_t)(kc2 * 32 + kk) * D_ + nb * 16 + mn];
            }
            oacc = __builtin_amdgcn_wmma_f32_16x16x32_bf16(
                     false, ap, false, bv, (short)0, oacc, false, false);
        }
        // store 16x16 context sub-tile
        const size_t crow0 = ((size_t)b * SQ_ + qbase) * D_;
#pragma unroll
        for (int r = 0; r < 8; ++r) {
            const int m = (lane < 16) ? r : (r + 8);
            const int n = lane & 15;
            ctx_out[crow0 + (size_t)m * D_ + nb * 16 + n] = oacc[r];
        }
    }
}

extern "C" void kernel_launch(void* const* d_in, const int* in_sizes, int n_in,
                              void* d_out, int out_size, void* d_ws, size_t ws_size,
                              hipStream_t stream) {
    const float* q = (const float*)d_in[0];
    const float* k = (const float*)d_in[1];
    const float* v = (const float*)d_in[2];

    float* ctx  = (float*)d_out;                       // [B,Sq,D] first
    float* attn = ctx + (size_t)B_ * SQ_ * D_;         // then [B,Sq,Sk]

    const size_t shmem = (size_t)(QTILE * SK_ + 256 + 16 + 16) * sizeof(float); // ~132 KB
    dim3 grid(SQ_ / QTILE, B_);
    dim3 block(256);
    attn_fwd_kernel<<<grid, block, shmem, stream>>>(q, k, v, ctx, attn);
}
